// GraphConvSparse_83811991814572
// MI455X (gfx1250) — matrix operand order repaired and checked by reading, not compiled, not run on hardware
//
#include <hip/hip_runtime.h>

// MI455X / gfx1250, wave32. D_IN = D_OUT = 128 fixed by the reference.
#define D 128
#define ROWTILES 5   // row-tiles (of 16) per GEMM block: 80 rows/block

typedef __attribute__((ext_vector_type(16))) _Float16 v16h;
typedef __attribute__((ext_vector_type(8)))  float    v8f;

// CDNA5 hardware transcendental v_tanh_f32 (confirmed lowering last round).
__device__ __forceinline__ float fast_tanh(float x) {
#if __has_builtin(__builtin_amdgcn_tanhf)
  return __builtin_amdgcn_tanhf(x);
#elif __has_builtin(__builtin_amdgcn_tanh_f32)
  return __builtin_amdgcn_tanh_f32(x);
#else
  return tanhf(x);
#endif
}

// ---------------------------------------------------------------------------
// Kernel 1: zero the accumulator (runs in d_out, which harness poisons)
// ---------------------------------------------------------------------------
__global__ __launch_bounds__(256) void zero_kernel(float4* __restrict__ p, long n4) {
  long i = (long)blockIdx.x * blockDim.x + threadIdx.x;
  long stride = (long)gridDim.x * blockDim.x;
  for (; i < n4; i += stride) p[i] = make_float4(0.f, 0.f, 0.f, 0.f);
}

// ---------------------------------------------------------------------------
// Kernel 2: edge scatter. Each wave32 owns 32 contiguous edges:
//   3 coalesced loads of (row, col, weight), then 32 shfl-broadcast steps.
//   Per step: one coalesced 512B float4 gather of x[c] (L2-resident: x=51MB
//   fits the 192MB L2) + 4 global_atomic_add_f32 into agg[r].
// ---------------------------------------------------------------------------
__global__ __launch_bounds__(256) void scatter_kernel(
    const float* __restrict__ x, const float* __restrict__ ew,
    const int* __restrict__ rows, const int* __restrict__ cols,
    float* __restrict__ agg, int n_edges) {
  const int wid  = (int)((blockIdx.x * (unsigned)blockDim.x + threadIdx.x) >> 5);
  const int lane = threadIdx.x & 31;
  const int nw   = (int)((gridDim.x * (unsigned)blockDim.x) >> 5);

  for (int base = wid * 32; base < n_edges; base += nw * 32) {
    const int idx = base + lane;
    int   r = 0, c = 0;
    float w = 0.f;
    if (idx < n_edges) {        // coalesced edge-stream loads (HBM)
      r = rows[idx];
      c = cols[idx];
      w = ew[idx];
    }
    int cnt = n_edges - base;
    if (cnt > 32) cnt = 32;
    for (int i = 0; i < cnt; ++i) {
      const int   ri = __shfl(r, i, 32);
      const int   ci = __shfl(c, i, 32);
      const float wi = __shfl(w, i, 32);
      const float4 v = ((const float4*)(x + (size_t)ci * D))[lane];
      float* dst = agg + (size_t)ri * D + lane * 4;
      atomicAdd(dst + 0, wi * v.x);
      atomicAdd(dst + 1, wi * v.y);
      atomicAdd(dst + 2, wi * v.z);
      atomicAdd(dst + 3, wi * v.w);
    }
  }
}

// ---------------------------------------------------------------------------
// Kernel 3: out = tanh(agg @ W^T), IN PLACE (io == agg == out).
// Block = 128 threads = 4 waves; block owns 16*ROWTILES rows exclusively.
// B fragments (full 128x128 W as f16, 2 N-tiles x 4 K-steps per wave) loaded
// ONCE per block into registers, reused across ROWTILES row-tiles.
// Per row-tile: load A, s_wait_loadcnt 0 + barrier (race-free in-place),
// 8x v_wmma_f32_16x16x32_f16, v_tanh_f32, store. Full blocks (the common
// case: 100000 % 80 == 0) take a block-uniform branch with unguarded stores.
// ---------------------------------------------------------------------------
__global__ __launch_bounds__(128) void gemm_tanh_kernel(
    float* __restrict__ io, const float* __restrict__ W, int n_nodes) {
  const int wave = threadIdx.x >> 5;
  const int lane = threadIdx.x & 31;
  const int half = lane >> 4;   // 0: lanes 0-15, 1: lanes 16-31
  const int l    = lane & 15;

  // ---- B fragments, loaded once: B[k][n] = W[n][k] --------------------
  // 16-bit B 32x16 layout (wave32): lane%16 = N, elems 0-7 -> K=kk+half*8+0..7, elems 8-15 -> K=kk+16+half*8+0..7.
  v16h bF[2][4];
#pragma unroll
  for (int t = 0; t < 2; ++t) {
    const int n0 = (wave * 2 + t) * 16;
    const float* wrow = W + (size_t)(n0 + l) * D;
#pragma unroll
    for (int i = 0; i < 4; ++i) {
      const int kk = i * 32;
#pragma unroll
      for (int e = 0; e < 16; ++e)
        bF[t][i][e] = (_Float16)wrow[kk + ((e >> 3) << 4) + half * 8 + (e & 7)];
    }
  }

  const int  mbase = blockIdx.x * (16 * ROWTILES);
  const bool full  = (mbase + 16 * ROWTILES) <= n_nodes;  // block-uniform

#pragma unroll 1
  for (int rt = 0; rt < ROWTILES; ++rt) {
    const int m0 = mbase + rt * 16;
    int mA = m0 + l;
    if (mA >= n_nodes) mA = n_nodes - 1;        // clamp, keep EXEC all-ones
    const float* arow = io + (size_t)mA * D;

    // A fragments (16-bit A 16x32 layout, wave32):
    //   half 0: elems 0-7 -> K=kk+0..7,  elems 8-15 -> K=kk+16..23
    //   half 1: elems 0-7 -> K=kk+8..15, elems 8-15 -> K=kk+24..31
    v16h aF[4];
#pragma unroll
    for (int i = 0; i < 4; ++i) {
      const int kk = i * 32;
#pragma unroll
      for (int e = 0; e < 8; ++e) {
        aF[i][e]     = (_Float16)arow[kk + half * 8 + e];
        aF[i][e + 8] = (_Float16)arow[kk + 16 + half * 8 + e];
      }
    }
    // All waves complete this tile's reads of io BEFORE any wave may pass
    // the barrier and overwrite the tile -> in-place is race-free.
    asm volatile("s_wait_loadcnt 0x0" ::: "memory");
    __syncthreads();

#pragma unroll
    for (int t = 0; t < 2; ++t) {
      const int n0 = (wave * 2 + t) * 16;
      v8f c = {};
#pragma unroll
      for (int i = 0; i < 4; ++i)
        // (neg_a, A, neg_b, B, c_mod, C, reuse_a, reuse_b)
        c = __builtin_amdgcn_wmma_f32_16x16x32_f16(false, aF[i], false,
                                                   bF[t][i], (short)0, c,
                                                   false, false);
      // C/D layout: VGPR v -> M = m0 + half*8 + v, lane l -> N = n0 + l
      float* obase = io + (size_t)(m0 + half * 8) * D + (n0 + l);
      if (full) {
        // Unguarded: one base address, 8 stores at immediate offsets (v*512B)
#pragma unroll
        for (int v = 0; v < 8; ++v) obase[(size_t)v * D] = fast_tanh(c[v]);
      } else {
#pragma unroll
        for (int v = 0; v < 8; ++v) {
          const int m = m0 + half * 8 + v;
          if (m < n_nodes) obase[(size_t)v * D] = fast_tanh(c[v]);
        }
      }
    }
  }
}

// ---------------------------------------------------------------------------
extern "C" void kernel_launch(void* const* d_in, const int* in_sizes, int n_in,
                              void* d_out, int out_size, void* d_ws, size_t ws_size,
                              hipStream_t stream) {
  const float* x  = (const float*)d_in[0];   // [N, 128] f32
  const float* W  = (const float*)d_in[1];   // [128, 128] f32
  const float* ew = (const float*)d_in[2];   // [E] f32
  const int*   ei = (const int*)d_in[3];     // [2, E] i32

  float* out = (float*)d_out;                // [N, 128] f32 (also the agg buffer)

  const int n_nodes = in_sizes[0] / D;
  const int n_edges = in_sizes[2];
  const int* rows = ei;            // edge_index[0]
  const int* cols = ei + n_edges;  // edge_index[1]

  // 1) zero accumulator (in d_out)
  long n4 = (long)n_nodes * (D / 4);
  zero_kernel<<<2048, 256, 0, stream>>>((float4*)out, n4);

  // 2) atomic scatter: one wave per 32 contiguous edges, 8 waves/block
  int waves  = (n_edges + 31) / 32;
  int blocks = (waves + 7) / 8;
  if (blocks > 262144) blocks = 262144;      // grid-stride loop covers rest
  scatter_kernel<<<blocks, 256, 0, stream>>>(x, ew, rows, cols, out, n_edges);

  // 3) in-place WMMA GEMM + tanh: one block per 80 rows (B held in regs)
  int gb = (n_nodes + 16 * ROWTILES - 1) / (16 * ROWTILES);
  gemm_tanh_kernel<<<gb, 128, 0, stream>>>(out, W, n_nodes);
}
